// Evo_Path_GNN_15169824489476
// MI455X (gfx1250) — matrix operand order repaired and checked by reading, not compiled, not run on hardware
//
#include <hip/hip_runtime.h>
#include <hip/hip_bf16.h>

typedef __attribute__((ext_vector_type(2)))  float  v2f;
typedef __attribute__((ext_vector_type(8)))  float  v8f;
typedef __attribute__((ext_vector_type(16))) __bf16 v16bf;

#define D 256

// ---------------------------------------------------------------------------
// Kernel 1: W = intsc_feat_fc @ messageNN^T   (256x256, one-off, tiny)
// W[i][j] = sum_k intsc[i][k] * msgNN[j][k]  (both contiguous row dots)
// ---------------------------------------------------------------------------
__global__ void fuse_w_kernel(const float* __restrict__ intsc,
                              const float* __restrict__ msgNN,
                              float* __restrict__ W) {
    const int i = blockIdx.x;    // row of W
    const int j = threadIdx.x;   // col of W
    const float* a = intsc + i * D;
    const float* b = msgNN + j * D;
    float acc = 0.0f;
    #pragma unroll 8
    for (int k = 0; k < D; ++k) acc += a[k] * b[k];
    W[i * D + j] = acc;
}

// ---------------------------------------------------------------------------
// Degree histogram (both endpoints of every edge)
// ---------------------------------------------------------------------------
__global__ void deg_init_kernel(float* __restrict__ deg) {
    if (threadIdx.x < 16) deg[threadIdx.x] = 0.0f;
}

__global__ void deg_accum_kernel(const int* __restrict__ el, int E,
                                 float* __restrict__ deg) {
    const int e = blockIdx.x * blockDim.x + threadIdx.x;
    if (e < E) {
        atomicAdd(&deg[el[e]],     1.0f);
        atomicAdd(&deg[el[E + e]], 1.0f);
    }
}

// ---------------------------------------------------------------------------
// Kernel 2: msg = edge_feat @ W   [E,256]x[256,256] f32 via V_WMMA_F32_16X16X4_F32
// One 16x16 output tile per wave; 8 waves / block; 64 K-steps of K=4.
// f32 A 16x4 layout: lanes0-15 hold (M=lane, K=k0+0/1), lanes16-31 (M=lane-16, K=k0+2/3)
// f32 B 4x16 layout: lanes0-15 hold (K=k0+0/1, N=lane), lanes16-31 (K=k0+2/3, N=lane-16)
// ---------------------------------------------------------------------------
__global__ void msg_gemm_kernel(const float* __restrict__ ef,
                                const float* __restrict__ W,
                                float* __restrict__ msg, int E) {
    const int wv     = threadIdx.x >> 5;
    const int lane   = threadIdx.x & 31;
    const int half   = lane >> 4;
    const int nn     = lane & 15;
    const int mtiles = (E + 15) >> 4;
    const int tileId = blockIdx.x * 8 + wv;
    if (tileId >= mtiles * 16) return;            // wave-uniform exit, EXEC stays full
    const int m0 = (tileId >> 4) * 16;
    const int n0 = (tileId & 15) * 16;

    int arow = m0 + nn;
    if (arow >= E) arow = E - 1;                  // clamp: tail rows read valid data
    const int n = n0 + nn;

    v8f c = {};
    for (int k0 = 0; k0 < D; k0 += 4) {
        const int kA = k0 + half * 2;
        const float2 av = *(const float2*)(ef + (size_t)arow * D + kA);
        v2f a, b;
        a[0] = av.x;               a[1] = av.y;
        b[0] = W[kA * D + n];      b[1] = W[(kA + 1) * D + n];
        c = __builtin_amdgcn_wmma_f32_16x16x4_f32(false, a, false, b,
                                                  (short)0, c, false, false);
    }
    #pragma unroll
    for (int r = 0; r < 8; ++r) {                 // VGPR r: M=r (lo lanes), M=8+r (hi lanes)
        const int row = m0 + r + half * 8;
        if (row < E) msg[(size_t)row * D + n0 + nn] = c[r];
    }
}

// ---------------------------------------------------------------------------
// Kernel 3: the sequential scan. Single workgroup, 512 threads = 16 waves.
// updateNN lives in LDS as bf16 pre-swizzled into WMMA B layout; each wave
// keeps its 8 K-slice B tiles (its 16-column N-tile) in VGPRs for the whole
// scan. Per update: lanes build vec (f32), convert to bf16, 8x WMMA bf16,
// row 0 of D is the new state slice.
// ---------------------------------------------------------------------------
#define SCAN_THREADS 512
#define OFF_UB   0          // 256*256 bf16 in B layout  = 131072 B
#define OFF_ST   131072     // state  10*256 f32         =  10240 B
#define OFF_NF   141312     // node_feat 10*256 f32      =  10240 B
#define OFF_DG   151552     // inv_deg 16 f32            =     64 B
#define OFF_VC   151616     // vec 256 bf16              =    512 B
#define SCAN_LDS 152192

__global__ void __launch_bounds__(SCAN_THREADS)
scan_kernel(const float* __restrict__ node_feat,
            const int*   __restrict__ el,
            const float* __restrict__ msg,
            const float* __restrict__ updNN,
            const float* __restrict__ deg,
            float* __restrict__ out,
            int E, int NNODES) {
    extern __shared__ char smem[];
    __bf16* Ub    = (__bf16*)(smem + OFF_UB);
    float*  state = (float*)(smem + OFF_ST);
    float*  nf    = (float*)(smem + OFF_NF);
    float*  idg   = (float*)(smem + OFF_DG);
    __bf16* vecbf = (__bf16*)(smem + OFF_VC);

    const int tid  = threadIdx.x;
    const int wv   = tid >> 5;
    const int lane = tid & 31;

    for (int i = tid; i < NNODES * D; i += SCAN_THREADS) {
        const float v = node_feat[i];
        state[i] = v;                 // evolving state
        nf[i]    = v;                 // frozen original node_feat
    }
    if (tid < 16) {
        const float d = (tid < NNODES) ? deg[tid] : 1.0f;
        idg[tid] = 1.0f / fmaxf(d, 1.0f);
    }
    // repack updateNN (k,n) f32 -> bf16 in WMMA 32x16 B layout, per K-slice k0
    // and N-tile w: chunk=(k0*16+w)*32+lane holds 16 contiguous bf16 (K pairs).
    for (int idx = tid; idx < D * D; idx += SCAN_THREADS) {
        const int k  = idx >> 8, n = idx & 255;
        const int k0 = k >> 5,  kk = k & 31;
        const int w  = n >> 4,  nc = n & 15;
        const int ln = nc + ((kk & 16) ? 16 : 0);
        const int chunk = (k0 * 16 + w) * 32 + ln;
        Ub[chunk * 16 + (kk & 15)] = (__bf16)updNN[k * D + n];
    }
    __syncthreads();

    // B tiles resident in VGPRs: 8 K-slices x 8 VGPRs = 64 VGPRs/lane
    v16bf bk[8];
    #pragma unroll
    for (int k0 = 0; k0 < 8; ++k0)
        bk[k0] = *(const v16bf*)(Ub + ((k0 * 16 + wv) * 32 + lane) * 16);

    float me = 0.0f;
    for (int e = 0; e < E; ++e) {
        const int s = el[e];
        const int t = el[E + e];
        if (tid < 256) me = msg[(size_t)e * D + tid];           // reused by both updates
        if (tid < 16 && e + 8 < E)                              // stream-ahead prefetch
            __builtin_prefetch(msg + (size_t)(e + 8) * D + tid * 16, 0, 1);

        #pragma unroll
        for (int u = 0; u < 2; ++u) {
            const int a = u ? t : s;      // node being updated
            const int b = u ? s : t;      // partner node (ORIGINAL node_feat)
            if (tid < 256) {
                const float v = state[a * D + tid] + idg[a] * me * nf[b * D + tid];
                vecbf[tid] = (__bf16)v;
            }
            __syncthreads();
            v8f c = {};
            #pragma unroll
            for (int k0 = 0; k0 < 8; ++k0) {
                // A k order chosen here: lo lanes K{0..15}, hi lanes K{16..31} (matches B repack);
                // every lane carries vec -> all 16 A rows equal vec, row 0 is the answer.
                union { uint4 q[2]; v16bf v; } A;
                const __bf16* p = vecbf + 32 * k0 + ((lane & 16) ? 16 : 0);
                A.q[0] = *(const uint4*)(p);
                A.q[1] = *(const uint4*)(p + 8);
                c = __builtin_amdgcn_wmma_f32_16x16x32_bf16(false, A.v, false, bk[k0],
                                                            (short)0, c, false, false);
            }
            if (lane < 16) state[a * D + wv * 16 + lane] = c[0];  // D row 0: VGPR0, lanes 0-15
            __syncthreads();
        }
    }
    for (int i = tid; i < NNODES * D; i += SCAN_THREADS) out[i] = state[i];
}

// ---------------------------------------------------------------------------
extern "C" void kernel_launch(void* const* d_in, const int* in_sizes, int n_in,
                              void* d_out, int out_size, void* d_ws, size_t ws_size,
                              hipStream_t stream) {
    const float* node_feat = (const float*)d_in[0];
    const float* edge_feat = (const float*)d_in[1];
    const int*   edge_list = (const int*)  d_in[2];
    const float* intsc     = (const float*)d_in[3];
    const float* messageNN = (const float*)d_in[4];
    const float* updateNN  = (const float*)d_in[5];
    float* out = (float*)d_out;

    const int NNODES = in_sizes[0] / D;
    const int E      = in_sizes[1] / D;

    float* W   = (float*)d_ws;        // 256*256 f32
    float* deg = W + D * D;           // 16 f32
    float* msg = deg + 64;            // E*256 f32

    fuse_w_kernel<<<D, D, 0, stream>>>(intsc, messageNN, W);
    deg_init_kernel<<<1, 64, 0, stream>>>(deg);
    deg_accum_kernel<<<(E + 255) / 256, 256, 0, stream>>>(edge_list, E, deg);

    const int mtiles = (E + 15) / 16;
    const int blocks = (mtiles * 16 + 7) / 8;     // one 16x16 tile per wave
    msg_gemm_kernel<<<blocks, 256, 0, stream>>>(edge_feat, W, msg, E);

    scan_kernel<<<1, SCAN_THREADS, SCAN_LDS, stream>>>(node_feat, edge_list, msg,
                                                       updateNN, deg, out, E, NNODES);
}
